// HarmonyTransformer_66760971649655
// MI455X (gfx1250) — compile-verified, matrix operand order checked by reading, not run on hardware
//
#include <hip/hip_runtime.h>

// Problem constants (from reference)
#define BB   8
#define SS   512
#define DD   512
#define HH   8
#define DHH  64
#define RPOS 1024   // padded distinct rel-pos rows (1023 used, row 1023 = zeros)

typedef __attribute__((ext_vector_type(16))) __bf16 v16bf;
typedef __attribute__((ext_vector_type(8)))  float  v8f;

union FragU { v16bf v; uint4 q[2]; };
union U4U16 { uint4 u; unsigned short s[8]; };

__device__ __forceinline__ unsigned short f2bf(float x) {
    unsigned int u = __float_as_uint(x);
    u = (u + 0x7fffu + ((u >> 16) & 1u)) >> 16;   // round-to-nearest-even
    return (unsigned short)u;
}
__device__ __forceinline__ float bf2f(unsigned short h) {
    return __uint_as_float(((unsigned int)h) << 16);
}

// Load a 16x32 bf16 WMMA A/B fragment from a row-major tile.
// Per CDNA5 ISA: lane L -> row (L&15); lanes 0-15 hold K {0..7,16..23},
// lanes 16-31 hold K {8..15,24..31}  => two contiguous 16B chunks.
__device__ __forceinline__ v16bf load_frag(const unsigned short* base, int ld,
                                           int rowbase, int kbase, int lane) {
    const int hl = (lane >> 4) & 1;
    const int r  = lane & 15;
    const unsigned short* p = base + (size_t)(rowbase + r) * ld + kbase + hl * 8;
    FragU f;
    f.q[0] = *(const uint4*)(p);
    f.q[1] = *(const uint4*)(p + 16);
    return f.v;
}

#define WMMA_BF16(A, Bm, C) \
    __builtin_amdgcn_wmma_f32_16x16x32_bf16(false, (A), false, (Bm), (short)0, (C), false, false)

// ---------------------------------------------------------------------------
// f32 -> bf16 conversion (grid-stride)
// ---------------------------------------------------------------------------
__global__ __launch_bounds__(256) void cvt_f32_to_bf16(const float* __restrict__ s,
                                                       unsigned short* __restrict__ d, int n) {
    for (int i = blockIdx.x * blockDim.x + threadIdx.x; i < n; i += gridDim.x * blockDim.x)
        d[i] = f2bf(s[i]);
}

// ---------------------------------------------------------------------------
// Gather the 1023 distinct sinusoidal rows out of pos_emb[S,S,D] -> bf16[1024,D]
// row j<=511:  pos_emb[511, j]   (distance j-511)
// row j>=512:  pos_emb[0, j-511] (distance j-511)
// row 1023: zeros (padding so the rr GEMM needs no bounds checks)
// ---------------------------------------------------------------------------
__global__ __launch_bounds__(256) void gather_pe(const float* __restrict__ pos,
                                                 unsigned short* __restrict__ pe) {
    const int j = blockIdx.x;
    unsigned short* drow = pe + (size_t)j * DD;
    if (j < 1023) {
        const int srow = (j < 512) ? (511 * SS + j) : (j - 511);
        const float* s = pos + (size_t)srow * DD;
        drow[threadIdx.x]       = f2bf(s[threadIdx.x]);
        drow[threadIdx.x + 256] = f2bf(s[threadIdx.x + 256]);
    } else {
        drow[threadIdx.x]       = 0;
        drow[threadIdx.x + 256] = 0;
    }
}

// ---------------------------------------------------------------------------
// Generic bf16 WMMA GEMM:  C[m,n] = sum_k A[m,k] * W[n,k]  (+ bias[n])
// Block tile 128x128, K-step 32, 8 waves (4 along M x 2 along N), each wave a
// 32x64 region = 2x4 v_wmma_f32_16x16x32_bf16 per K-step. Double-buffered LDS:
// next tile's global loads + ds stores overlap current tile's WMMAs; one
// barrier per K-step. Optional z-batching (used for the per-head BD GEMM):
//   A += (z/8)*sAhi + (z%8)*sAlo ;  W += (z%8)*sWlo ;  outputs += z*sCz
// Epilogue: up to two bf16 outputs (each with an extra per-column bias) and/or
// one f32 output with optional residual add. All dims divide the tiles.
// ---------------------------------------------------------------------------
__global__ __launch_bounds__(256) void gemm_bf16_tn(
    const unsigned short* __restrict__ A, int lda,
    const unsigned short* __restrict__ W, int ldw,
    const float* __restrict__ bias, int K,
    unsigned short* __restrict__ out1, const float* __restrict__ extra1,
    unsigned short* __restrict__ out2, const float* __restrict__ extra2,
    float* __restrict__ outf, const float* __restrict__ resid, int ldc,
    size_t sAhi, int sAlo, int sWlo, size_t sCz) {
    __shared__ unsigned short As[2][128 * 32];
    __shared__ unsigned short Ws[2][128 * 32];

    const int z = blockIdx.z, zb = z >> 3, zh = z & 7;
    A += (size_t)zb * sAhi + (size_t)zh * sAlo;
    W += (size_t)zh * sWlo;
    const size_t co = (size_t)z * sCz;
    if (out1) out1 += co;
    if (out2) out2 += co;
    if (outf) outf += co;
    if (resid) resid += co;

    const int m0 = blockIdx.y * 128, n0 = blockIdx.x * 128;
    const int t = threadIdx.x, lane = t & 31, w = t >> 5;
    const int wm = w & 3, wn = w >> 2;   // wave covers rows wm*32..+32, cols wn*64..+64

    v8f acc[2][4];
#pragma unroll
    for (int i = 0; i < 2; ++i)
#pragma unroll
        for (int j = 0; j < 4; ++j)
#pragma unroll
            for (int e = 0; e < 8; ++e) acc[i][j][e] = 0.f;

    const int rowL = t >> 1;
    const int segL = (t & 1) * 16;
    const unsigned short* ga = A + (size_t)(m0 + rowL) * lda + segL;
    const unsigned short* gw = W + (size_t)(n0 + rowL) * ldw + segL;

    // prologue: stage K-tile 0 into buffer 0
    uint4 ra0 = *(const uint4*)(ga), ra1 = *(const uint4*)(ga + 8);
    uint4 rw0 = *(const uint4*)(gw), rw1 = *(const uint4*)(gw + 8);
    *(uint4*)&As[0][rowL * 32 + segL]     = ra0;
    *(uint4*)&As[0][rowL * 32 + segL + 8] = ra1;
    *(uint4*)&Ws[0][rowL * 32 + segL]     = rw0;
    *(uint4*)&Ws[0][rowL * 32 + segL + 8] = rw1;
    __syncthreads();

    int buf = 0;
    for (int k0 = 0; k0 < K; k0 += 32, buf ^= 1) {
        const bool more = (k0 + 32) < K;
        if (more) {   // prefetch next tile into registers (overlaps WMMAs below)
            ra0 = *(const uint4*)(ga + k0 + 32);
            ra1 = *(const uint4*)(ga + k0 + 40);
            rw0 = *(const uint4*)(gw + k0 + 32);
            rw1 = *(const uint4*)(gw + k0 + 40);
        }

        v16bf af0 = load_frag(As[buf], 32, wm * 32,      0, lane);
        v16bf af1 = load_frag(As[buf], 32, wm * 32 + 16, 0, lane);
#pragma unroll
        for (int fn = 0; fn < 4; ++fn) {
            v16bf bfr = load_frag(Ws[buf], 32, wn * 64 + fn * 16, 0, lane);
            acc[0][fn] = WMMA_BF16(af0, bfr, acc[0][fn]);
            acc[1][fn] = WMMA_BF16(af1, bfr, acc[1][fn]);
        }

        if (more) {   // write prefetched tile to the other buffer
            *(uint4*)&As[buf ^ 1][rowL * 32 + segL]     = ra0;
            *(uint4*)&As[buf ^ 1][rowL * 32 + segL + 8] = ra1;
            *(uint4*)&Ws[buf ^ 1][rowL * 32 + segL]     = rw0;
            *(uint4*)&Ws[buf ^ 1][rowL * 32 + segL + 8] = rw1;
        }
        __syncthreads();
    }

    const int hl = (lane >> 4) & 1, cn = lane & 15;
#pragma unroll
    for (int fm = 0; fm < 2; ++fm)
#pragma unroll
        for (int fn = 0; fn < 4; ++fn)
#pragma unroll
            for (int e = 0; e < 8; ++e) {
                const int gm = m0 + wm * 32 + fm * 16 + hl * 8 + e;
                const int gn = n0 + wn * 64 + fn * 16 + cn;
                float val = acc[fm][fn][e];
                if (bias) val += bias[gn];
                const size_t o = (size_t)gm * ldc + gn;
                if (out1) out1[o] = f2bf(val + (extra1 ? extra1[gn] : 0.f));
                if (out2) out2[o] = f2bf(val + (extra2 ? extra2[gn] : 0.f));
                if (outf) outf[o] = val + (resid ? resid[o] : 0.f);
            }
}

// ---------------------------------------------------------------------------
// Flash attention with precomputed rel-shift scores.
// Grid: (S/64, H, B). Block 256 = 8 waves; wave w: rows (w>>1)*16, cols (w&1)*32.
// scores = (qu_tile @ kh_tileT + BD[b,h,q, k+511-q]) * (1/sqrt(DH)); online
// softmax; O += P @ V. V gets its own LDS buffer (staged transposed) so its
// global loads are issued with the K loads and overlap the score/softmax
// phase; 4 barriers per k-tile.
// ---------------------------------------------------------------------------
__global__ __launch_bounds__(256) void attn_kernel(
    const unsigned short* __restrict__ qu,   // [B,S,D] bf16: q@WqT + bq + u_bias
    const unsigned short* __restrict__ kh,   // [B,S,D] bf16
    const unsigned short* __restrict__ vh,   // [B,S,D] bf16
    const unsigned short* __restrict__ BD,   // [B*H, S, RPOS] bf16
    unsigned short* __restrict__ attn)       // [B,S,D] bf16
{
    __shared__ unsigned short qu_s[64 * 64];
    __shared__ unsigned short k_s[64 * 64];
    __shared__ unsigned short vt_s[64 * 64];   // V^T : vt_s[d][key]
    __shared__ float          sc_s[64 * 64];
    __shared__ unsigned short p_s[64 * 64];
    __shared__ float mrow[64], lrow[64], crow[64];

    const int q0 = blockIdx.x * 64;
    const int h  = blockIdx.y;
    const int b  = blockIdx.z;
    const int t = threadIdx.x, lane = t & 31, w = t >> 5;
    const int wm = w >> 1;                 // 0..3 -> rows wm*16
    const int wn = w & 1;                  // 0..1 -> cols wn*32
    const int hl = (lane >> 4) & 1, cn = lane & 15;

    const int rowT = t >> 2, segT = (t & 3) * 16;   // 64x64 bf16 tile staging map

    {   // Q tile (rows q0.., head slice h*64..)
        const unsigned short* g = qu + (size_t)(b * SS + q0 + rowT) * DD + h * DHH + segT;
        *(uint4*)&qu_s[rowT * 64 + segT]     = *(const uint4*)(g);
        *(uint4*)&qu_s[rowT * 64 + segT + 8] = *(const uint4*)(g + 8);
    }
    if (t < 64) { mrow[t] = -1e30f; lrow[t] = 0.f; }

    v8f oacc[2];
#pragma unroll
    for (int fn = 0; fn < 2; ++fn)
#pragma unroll
        for (int e = 0; e < 8; ++e) oacc[fn][e] = 0.f;

    const unsigned short* BDb = BD + (size_t)(b * HH + h) * SS * RPOS;

    for (int k0 = 0; k0 < SS; k0 += 64) {
        // ---- issue K and V global loads together ----
        const unsigned short* gk = kh + (size_t)(b * SS + k0 + rowT) * DD + h * DHH + segT;
        const uint4 kreg0 = *(const uint4*)(gk);
        const uint4 kreg1 = *(const uint4*)(gk + 8);
        U4U16 vreg[2];
#pragma unroll
        for (int c = 0; c < 2; ++c) {
            const int f = t + c * 256;               // 0..511 : (key, 8-wide d chunk)
            const int key = f >> 3, d8 = (f & 7) * 8;
            vreg[c].u = *(const uint4*)(vh + (size_t)(b * SS + k0 + key) * DD + h * DHH + d8);
        }
        // K tile -> k_s[key][d]   (prev score reads of k_s are behind >=1 barrier)
        *(uint4*)&k_s[rowT * 64 + segT]     = kreg0;
        *(uint4*)&k_s[rowT * 64 + segT + 8] = kreg1;
        __syncthreads();   // A: k_s ready; prev-iter PV reads of vt_s/p_s done

        // V tile -> vt_s[d][key] (transposed store; overlaps score WMMAs)
#pragma unroll
        for (int c = 0; c < 2; ++c) {
            const int f = t + c * 256;
            const int key = f >> 3, d8 = (f & 7) * 8;
#pragma unroll
            for (int j = 0; j < 8; ++j) vt_s[(d8 + j) * 64 + key] = vreg[c].s[j];
        }

        // ---- scores: qu_tile (64x64) @ kh_tile^T (64x64) ----
        v8f sacc[2];
#pragma unroll
        for (int fn = 0; fn < 2; ++fn)
#pragma unroll
            for (int e = 0; e < 8; ++e) sacc[fn][e] = 0.f;
#pragma unroll
        for (int kk = 0; kk < 64; kk += 32) {
            v16bf af = load_frag(qu_s, 64, wm * 16, kk, lane);
#pragma unroll
            for (int fn = 0; fn < 2; ++fn) {
                v16bf bfr = load_frag(k_s, 64, wn * 32 + fn * 16, kk, lane);
                sacc[fn] = WMMA_BF16(af, bfr, sacc[fn]);
            }
        }
#pragma unroll
        for (int fn = 0; fn < 2; ++fn)
#pragma unroll
            for (int e = 0; e < 8; ++e)
                sc_s[(wm * 16 + hl * 8 + e) * 64 + wn * 32 + fn * 16 + cn] = sacc[fn][e];
        __syncthreads();   // B: sc_s ready, vt_s stores done

        // ---- add shifted BD slice, scale by 1/sqrt(DH) (lane-coalesced) ----
#pragma unroll
        for (int i = 0; i < 16; ++i) {
            const int f = t + i * 256;               // 0..4095
            const int row = f >> 6, col = f & 63;
            const int q = q0 + row;
            const int j = k0 + col + 511 - q;        // always in [0,1022]
            sc_s[f] = (sc_s[f] + bf2f(BDb[(size_t)q * RPOS + j])) * 0.125f;
        }
        __syncthreads();   // C

        // ---- row-wise online softmax (one thread per row) ----
        if (t < 64) {
            const float* sr = &sc_s[t * 64];
            float mold = mrow[t], mnew = mold;
            for (int c = 0; c < 64; ++c) mnew = fmaxf(mnew, sr[c]);
            const float corr = __expf(mold - mnew);
            float sum = 0.f;
            for (int c = 0; c < 64; ++c) {
                const float p = __expf(sr[c] - mnew);
                sum += p;
                p_s[t * 64 + c] = f2bf(p);
            }
            lrow[t] = lrow[t] * corr + sum;
            mrow[t] = mnew;
            crow[t] = corr;
        }
        __syncthreads();   // D: p_s / crow ready

        // ---- rescale O accumulators, then O += P (64x64) @ V (64x64) ----
#pragma unroll
        for (int fn = 0; fn < 2; ++fn)
#pragma unroll
            for (int e = 0; e < 8; ++e) oacc[fn][e] *= crow[wm * 16 + hl * 8 + e];
#pragma unroll
        for (int kk = 0; kk < 64; kk += 32) {
            v16bf af = load_frag(p_s, 64, wm * 16, kk, lane);
#pragma unroll
            for (int fn = 0; fn < 2; ++fn) {
                v16bf bfr = load_frag(vt_s, 64, wn * 32 + fn * 16, kk, lane);
                oacc[fn] = WMMA_BF16(af, bfr, oacc[fn]);
            }
        }
        // next iteration's barrier A orders these reads vs. vt_s/p_s rewrites
    }

    // ---- normalize and store [B,S,H,DH] -> [B,S,D] ----
#pragma unroll
    for (int fn = 0; fn < 2; ++fn)
#pragma unroll
        for (int e = 0; e < 8; ++e) {
            const int row = wm * 16 + hl * 8 + e;
            const int col = wn * 32 + fn * 16 + cn;
            const float val = oacc[fn][e] / lrow[row];
            attn[(size_t)(b * SS + q0 + row) * DD + h * DHH + col] = f2bf(val);
        }
}

// ---------------------------------------------------------------------------
// LayerNorm over last dim (512) ; one block per row
// ---------------------------------------------------------------------------
__global__ __launch_bounds__(256) void ln_kernel(const float* __restrict__ y,
                                                 const float* __restrict__ g,
                                                 const float* __restrict__ be,
                                                 float* __restrict__ out) {
    __shared__ float red[256];
    __shared__ float mu_s, inv_s;
    const int row = blockIdx.x, t = threadIdx.x;
    const float* yr = y + (size_t)row * DD;
    const float a = yr[t], c = yr[t + 256];

    red[t] = a + c;
    __syncthreads();
    for (int s = 128; s > 0; s >>= 1) { if (t < s) red[t] += red[t + s]; __syncthreads(); }
    if (t == 0) mu_s = red[0] * (1.f / DD);
    __syncthreads();
    const float mu = mu_s;
    const float d0 = a - mu, d1 = c - mu;
    red[t] = d0 * d0 + d1 * d1;
    __syncthreads();
    for (int s = 128; s > 0; s >>= 1) { if (t < s) red[t] += red[t + s]; __syncthreads(); }
    if (t == 0) inv_s = rsqrtf(red[0] * (1.f / DD) + 1e-5f);
    __syncthreads();
    const float inv = inv_s;
    out[(size_t)row * DD + t]       = d0 * inv * g[t] + be[t];
    out[(size_t)row * DD + t + 256] = d1 * inv * g[t + 256] + be[t + 256];
}

// ---------------------------------------------------------------------------
extern "C" void kernel_launch(void* const* d_in, const int* in_sizes, int n_in,
                              void* d_out, int out_size, void* d_ws, size_t ws_size,
                              hipStream_t stream) {
    (void)in_sizes; (void)n_in; (void)out_size; (void)ws_size;
    const float* q   = (const float*)d_in[0];
    const float* k   = (const float*)d_in[1];
    const float* v   = (const float*)d_in[2];
    const float* pos = (const float*)d_in[3];
    const float* Wq  = (const float*)d_in[4];  const float* bq = (const float*)d_in[5];
    const float* Wk  = (const float*)d_in[6];  const float* bk = (const float*)d_in[7];
    const float* Wv  = (const float*)d_in[8];  const float* bv = (const float*)d_in[9];
    const float* Wr  = (const float*)d_in[10]; const float* br = (const float*)d_in[11];
    const float* Wo  = (const float*)d_in[12]; const float* bo = (const float*)d_in[13];
    const float* ub  = (const float*)d_in[14];
    const float* vb  = (const float*)d_in[15];
    const float* lg  = (const float*)d_in[16];
    const float* lb  = (const float*)d_in[17];

    char* ws = (char*)d_ws;
    size_t off = 0;
    auto take = [&](size_t bytes) -> char* {
        char* p = ws + off;
        off = (off + bytes + 255) & ~(size_t)255;
        return p;
    };

    const size_t MSD = (size_t)BB * SS * DD;   // 2,097,152 elements
    unsigned short* q_bf   = (unsigned short*)take(MSD * 2);
    unsigned short* k_bf   = (unsigned short*)take(MSD * 2);
    unsigned short* v_bf   = (unsigned short*)take(MSD * 2);
    unsigned short* Wq_bf  = (unsigned short*)take((size_t)DD * DD * 2);
    unsigned short* Wk_bf  = (unsigned short*)take((size_t)DD * DD * 2);
    unsigned short* Wv_bf  = (unsigned short*)take((size_t)DD * DD * 2);
    unsigned short* Wr_bf  = (unsigned short*)take((size_t)DD * DD * 2);
    unsigned short* Wo_bf  = (unsigned short*)take((size_t)DD * DD * 2);
    unsigned short* pe_bf  = (unsigned short*)take((size_t)RPOS * DD * 2);
    unsigned short* rr_bf  = (unsigned short*)take((size_t)RPOS * DD * 2);
    unsigned short* qu_bf  = (unsigned short*)take(MSD * 2);   // qh + u_bias
    unsigned short* qvb_bf = (unsigned short*)take(MSD * 2);   // qh + v_bias
    unsigned short* kh_bf  = (unsigned short*)take(MSD * 2);
    unsigned short* vh_bf  = (unsigned short*)take(MSD * 2);
    unsigned short* BD_bf  = (unsigned short*)take((size_t)BB * HH * SS * RPOS * 2); // 64 MiB
    unsigned short* at_bf  = (unsigned short*)take(MSD * 2);
    float*          y_f32  = (float*)take(MSD * 4);

    const int nIn = (int)MSD, nW = DD * DD;
    cvt_f32_to_bf16<<<dim3((nIn + 1023) / 1024), 256, 0, stream>>>(q, q_bf, nIn);
    cvt_f32_to_bf16<<<dim3((nIn + 1023) / 1024), 256, 0, stream>>>(k, k_bf, nIn);
    cvt_f32_to_bf16<<<dim3((nIn + 1023) / 1024), 256, 0, stream>>>(v, v_bf, nIn);
    cvt_f32_to_bf16<<<dim3((nW + 1023) / 1024), 256, 0, stream>>>(Wq, Wq_bf, nW);
    cvt_f32_to_bf16<<<dim3((nW + 1023) / 1024), 256, 0, stream>>>(Wk, Wk_bf, nW);
    cvt_f32_to_bf16<<<dim3((nW + 1023) / 1024), 256, 0, stream>>>(Wv, Wv_bf, nW);
    cvt_f32_to_bf16<<<dim3((nW + 1023) / 1024), 256, 0, stream>>>(Wr, Wr_bf, nW);
    cvt_f32_to_bf16<<<dim3((nW + 1023) / 1024), 256, 0, stream>>>(Wo, Wo_bf, nW);
    gather_pe<<<dim3(RPOS), 256, 0, stream>>>(pos, pe_bf);

    const dim3 gProj(DD / 128, (BB * SS) / 128);   // (4, 32)
    // q projection: emit (qh+bq+u_bias) and (qh+bq+v_bias) as bf16 in one pass
    gemm_bf16_tn<<<gProj, 256, 0, stream>>>(q_bf, DD, Wq_bf, DD, bq, DD,
                                            qu_bf, ub, qvb_bf, vb,
                                            nullptr, nullptr, DD, 0, 0, 0, 0);
    gemm_bf16_tn<<<gProj, 256, 0, stream>>>(k_bf, DD, Wk_bf, DD, bk, DD,
                                            kh_bf, nullptr, nullptr, nullptr,
                                            nullptr, nullptr, DD, 0, 0, 0, 0);
    gemm_bf16_tn<<<gProj, 256, 0, stream>>>(v_bf, DD, Wv_bf, DD, bv, DD,
                                            vh_bf, nullptr, nullptr, nullptr,
                                            nullptr, nullptr, DD, 0, 0, 0, 0);
    // rr = pe_rows @ Wr.T + br   (M=1024 incl. zero pad row)
    gemm_bf16_tn<<<dim3(DD / 128, RPOS / 128), 256, 0, stream>>>(
        pe_bf, DD, Wr_bf, DD, br, DD,
        rr_bf, nullptr, nullptr, nullptr, nullptr, nullptr, DD, 0, 0, 0, 0);

    // BD[b,h,q,j] = (qh+v_bias)[b,q,h,:] . rr[j,h,:]
    // single batched launch, z = b*H + h  (A += (z/8)*S*D + (z%8)*DH, W += (z%8)*DH)
    gemm_bf16_tn<<<dim3(RPOS / 128, SS / 128, BB * HH), 256, 0, stream>>>(
        qvb_bf, DD, rr_bf, DD, nullptr, DHH,
        BD_bf, nullptr, nullptr, nullptr, nullptr, nullptr, RPOS,
        (size_t)SS * DD, DHH, DHH, (size_t)SS * RPOS);

    attn_kernel<<<dim3(SS / 64, HH, BB), 256, 0, stream>>>(qu_bf, kh_bf, vh_bf, BD_bf, at_bf);

    // y = residual(q) + attn @ Wo.T + bo   (f32)
    gemm_bf16_tn<<<gProj, 256, 0, stream>>>(at_bf, DD, Wo_bf, DD, bo, DD,
                                            nullptr, nullptr, nullptr, nullptr,
                                            y_f32, q, DD, 0, 0, 0, 0);

    ln_kernel<<<dim3(BB * SS), 256, 0, stream>>>(y_f32, lg, lb, (float*)d_out);
}
